// BiLSTMTagger_52304111730764
// MI455X (gfx1250) — matrix operand order repaired and checked by reading, not compiled, not run on hardware
//
#include <hip/hip_runtime.h>
#include <hip/hip_bf16.h>
#include <math.h>

typedef __attribute__((ext_vector_type(16))) _Float16 v16h;
typedef __attribute__((ext_vector_type(8)))  float    v8f;
typedef _Float16 half_t;

// ---------------------------------------------------------------------------
// Model dimensions
// ---------------------------------------------------------------------------
#define BB   16
#define LL   128
#define BL   2048      // B*L
#define HH   300
#define H4   1200      // 4*H
#define KX   928       // sent_dim 917 padded to mult of 32
#define D2   608       // 2H=600 padded to mult of 32 (and 16 for N-tiles)
#define DHS  1216      // 4H=1200 padded
#define KWH  320       // H=300 padded
#define RR   20
#define RPAD 32
#define RDD  128
#define NDEP 64        // 50 padded

__device__ inline float sigf(float x) { return 1.0f / (1.0f + expf(-x)); }

// ---------------------------------------------------------------------------
// WMMA fragment loaders (wave32; contiguous k = 16*(lane>>4) + s, same order for A and B)
// ap = A + (mr+lr)*lda + (lane>>4)*16 ; wp = W + (nb+lr)*ldw + (lane>>4)*16
// Contiguous 16B runs -> compiler merges into global_load_b128 / ds_load_b128.
// ---------------------------------------------------------------------------
__device__ inline v16h load_fa(const half_t* __restrict__ ap, int kb) {
    v16h a;
#pragma unroll
    for (int v = 0; v < 4; ++v) { a[2*v] = ap[kb + 2*v]; a[2*v+1] = ap[kb + 2*v+1]; }
#pragma unroll
    for (int v = 0; v < 4; ++v) { a[8+2*v] = ap[kb + 8 + 2*v]; a[9+2*v] = ap[kb + 8 + 2*v+1]; }
    return a;
}
__device__ inline v16h load_fb(const half_t* __restrict__ wp, int kb) {
    v16h b;
#pragma unroll
    for (int v = 0; v < 8; ++v) { b[2*v] = wp[kb + 2*v]; b[2*v+1] = wp[kb + 2*v+1]; }
    return b;
}

// ---------------------------------------------------------------------------
// Generic WMMA GEMM: D[M][N] = act( A[M][K](f16) * W[N][K]^T(f16) + Cin + bias )
// One wave computes a 16x(16*JMAX) strip, reusing the A fragment JMAX times.
// JMAX is a template constant -> the j-loop is fully unrolled and branch-free,
// so accumulators stay pinned in VGPRs. Host splits N into a JMAX=4 launch
// plus one remainder launch (JMAX=1..3) at column offset nbBase.
// act: 0=none 1=relu 2=tanh
// ---------------------------------------------------------------------------
template<int JMAX>
__global__ __launch_bounds__(256) void k_gemm(
    const half_t* __restrict__ A, int lda, long long sA,
    const half_t* __restrict__ W, int ldw, long long sW,
    const float* __restrict__ bias,
    const float* __restrict__ Cin, int ldci, long long sCi,
    float* __restrict__ Cout, int ldc, long long sC,
    half_t* __restrict__ Hout, int ldh, long long sH,
    int M, int K, int act, int nbBase, int nGroups)
{
    long long batch = blockIdx.y;
    A += batch * sA;  W += batch * sW;
    const int lane = threadIdx.x & 31;
    const int wib  = __builtin_amdgcn_readfirstlane((int)threadIdx.x >> 5);
    const int waveId = (int)blockIdx.x * 8 + wib;

    const int nTiles = (M >> 4) * nGroups;
    if (waveId >= nTiles) return;
    const int mr  = (waveId / nGroups) << 4;
    const int nb0 = nbBase + (waveId % nGroups) * (JMAX << 4);
    const int hh = lane >> 4, lr = lane & 15;

    v8f c[JMAX];
    if (Cin) {
        const float* ci = Cin + batch * sCi;
#pragma unroll
        for (int j = 0; j < JMAX; ++j)
#pragma unroll
            for (int r = 0; r < 8; ++r)
                c[j][r] = ci[(long long)(mr + r + 8*hh) * ldci + nb0 + (j<<4) + lr];
    } else {
#pragma unroll
        for (int j = 0; j < JMAX; ++j)
#pragma unroll
            for (int r = 0; r < 8; ++r) c[j][r] = 0.0f;
    }

    const half_t* ap  = A + (long long)(mr + lr) * lda + hh * 16;
    const half_t* wpb = W + (long long)(nb0 + lr) * ldw + hh * 16;
    for (int kb = 0; kb < K; kb += 32) {
        __builtin_prefetch(ap + kb + 256, 0, 1);   // global_prefetch_b8 (activation stream)
        v16h a = load_fa(ap, kb);
#pragma unroll
        for (int j = 0; j < JMAX; ++j) {
            v16h b = load_fb(wpb + (long long)(j << 4) * ldw, kb);
            c[j] = __builtin_amdgcn_wmma_f32_16x16x32_f16(false, a, false, b,
                                                          (short)0, c[j], false, false);
        }
    }

#pragma unroll
    for (int j = 0; j < JMAX; ++j) {
#pragma unroll
        for (int r = 0; r < 8; ++r) {
            float v = c[j][r] + (bias ? bias[nb0 + (j<<4) + lr] : 0.0f);
            if (act == 1) v = v > 0.0f ? v : 0.0f;
            else if (act == 2) v = tanhf(v);
            long long row = mr + r + 8*hh;
            if (Cout) (Cout + batch * sC)[row * ldc + nb0 + (j<<4) + lr] = v;
            if (Hout) (Hout + batch * sH)[row * ldh + nb0 + (j<<4) + lr] = (half_t)v;
        }
    }
}

// ---------------------------------------------------------------------------
// Persistent LSTM scan: one block per direction, h/c state in LDS,
// gate GEMM g = xp[:,t,:] + h @ whh^T via WMMA (75 N-tiles over 32 waves).
// Writes f16 output into concat buffer (cols dirOff..dirOff+299).
// ---------------------------------------------------------------------------
__global__ __launch_bounds__(1024) void k_lstm_scan(
    const float* __restrict__ xpF, const float* __restrict__ xpB,
    const half_t* __restrict__ whhF, const half_t* __restrict__ whhB,
    half_t* __restrict__ outH, int ldh)
{
    const int dir = blockIdx.x;
    const float*  xp  = dir ? xpB  : xpF;
    const half_t* whh = dir ? whhB : whhF;
    const int dirOff = dir ? HH : 0;

    __shared__ half_t h_s[16 * KWH];     // hidden state, cols 300..319 stay 0
    __shared__ half_t c_s[16 * 304];     // cell state
    __shared__ half_t g_s[16 * H4];      // gate pre-activations

    const int tid = threadIdx.x;
    for (int i = tid; i < 16 * KWH; i += blockDim.x) h_s[i] = (half_t)0.0f;
    for (int i = tid; i < 16 * 304; i += blockDim.x) c_s[i] = (half_t)0.0f;
    __syncthreads();

    const int lane = tid & 31;
    const int wid  = __builtin_amdgcn_readfirstlane(tid >> 5);
    const int hh = lane >> 4, lr = lane & 15;

    for (int t = 0; t < LL; ++t) {
        const int tt = dir ? (LL - 1 - t) : t;
        for (int tn = wid; tn < 75; tn += 32) {
            const int nb = tn << 4;
            v8f c;
#pragma unroll
            for (int r = 0; r < 8; ++r)
                c[r] = xp[((long long)(r + 8*hh) * LL + tt) * H4 + nb + lr];
            const half_t* ap = h_s + lr * KWH + hh * 16;
            const half_t* wp = whh + (long long)(nb + lr) * KWH + hh * 16;
            for (int kb = 0; kb < KWH; kb += 32) {
                v16h a = load_fa(ap, kb);
                v16h b = load_fb(wp, kb);
                c = __builtin_amdgcn_wmma_f32_16x16x32_f16(false, a, false, b,
                                                           (short)0, c, false, false);
            }
#pragma unroll
            for (int r = 0; r < 8; ++r)
                g_s[(r + 8*hh) * H4 + nb + lr] = (half_t)c[r];
        }
        __syncthreads();
        for (int i = tid; i < 16 * HH; i += blockDim.x) {
            const int b = i / HH, j = i % HH;
            float gi = (float)g_s[b * H4 + j];
            float gf = (float)g_s[b * H4 + HH + j];
            float gg = (float)g_s[b * H4 + 2*HH + j];
            float go = (float)g_s[b * H4 + 3*HH + j];
            float cc = sigf(gf) * (float)c_s[b * 304 + j] + sigf(gi) * tanhf(gg);
            float hv = sigf(go) * tanhf(cc);
            c_s[b * 304 + j] = (half_t)cc;
            h_s[b * KWH + j] = (half_t)hv;
            outH[((long long)b * LL + tt) * ldh + dirOff + j] = (half_t)hv;
        }
        __syncthreads();
    }
}

// ---------------------------------------------------------------------------
// Elementwise / data-movement kernels
// ---------------------------------------------------------------------------
__global__ void k_cvt(const float* __restrict__ src, int rows, int cols,
                      half_t* __restrict__ dst, int ldd, int rpad)
{
    long long n = (long long)rpad * ldd;
    for (long long i = (long long)blockIdx.x * blockDim.x + threadIdx.x; i < n;
         i += (long long)gridDim.x * blockDim.x) {
        int r = (int)(i / ldd), cI = (int)(i % ldd);
        float v = (r < rows && cI < cols) ? src[(long long)r * cols + cI] : 0.0f;
        dst[i] = (half_t)v;
    }
}

__global__ void k_padf32(const float* __restrict__ src, int n, float* __restrict__ dst, int npad)
{
    for (int i = blockIdx.x * blockDim.x + threadIdx.x; i < npad; i += gridDim.x * blockDim.x)
        dst[i] = (i < n) ? src[i] : 0.0f;
}

__global__ void k_embed(const int* __restrict__ sent, const int* __restrict__ psent,
                        const int* __restrict__ pos, const int* __restrict__ lemma,
                        const float* __restrict__ region,
                        const float* __restrict__ we, const float* __restrict__ wf,
                        const float* __restrict__ pe, const float* __restrict__ le,
                        half_t* __restrict__ X)
{
    long long n = (long long)BL * KX;
    for (long long i = (long long)blockIdx.x * blockDim.x + threadIdx.x; i < n;
         i += (long long)gridDim.x * blockDim.x) {
        int row = (int)(i / KX), c = (int)(i % KX);
        float v;
        if      (c < 300) v = we[(long long)sent[row]  * 300 + c];
        else if (c < 600) v = wf[(long long)psent[row] * 300 + (c - 300)];
        else if (c < 616) v = pe[(long long)pos[row]   * 16  + (c - 600)];
        else if (c < 916) v = le[(long long)lemma[row] * 300 + (c - 616)];
        else if (c == 916) v = region[row];
        else v = 0.0f;
        X[i] = (half_t)v;
    }
}

__global__ void k_gather_dep(const half_t* __restrict__ bfe, const int* __restrict__ heads,
                             half_t* __restrict__ ce)
{
    long long n = (long long)BL * D2;
    for (long long i = (long long)blockIdx.x * blockDim.x + threadIdx.x; i < n;
         i += (long long)gridDim.x * blockDim.x) {
        int row = (int)(i / D2), c = (int)(i % D2);
        int b = row >> 7;
        int hd = heads[row];
        int src = min(max(hd - 1, 0), LL - 1);
        half_t v = (half_t)0.0f;
        if (hd > 0 && c < 600) v = bfe[((long long)(b * LL) + src) * D2 + c];
        ce[i] = v;
    }
}

__global__ void k_predcat(const half_t* __restrict__ h3, const int* __restrict__ tgt,
                          half_t* __restrict__ hs)
{
    long long n = (long long)BL * DHS;
    for (long long i = (long long)blockIdx.x * blockDim.x + threadIdx.x; i < n;
         i += (long long)gridDim.x * blockDim.x) {
        int row = (int)(i / DHS), c = (int)(i % DHS);
        int b = row >> 7;
        half_t v = (half_t)0.0f;
        if (c < 600) v = h3[(long long)row * D2 + c];
        else if (c < 1200) v = h3[((long long)b * LL + tgt[b]) * D2 + (c - 600)];
        hs[i] = v;
    }
}

__global__ void k_rolebuild(const int* __restrict__ voc, const int* __restrict__ frames,
                            const float* __restrict__ remb, const float* __restrict__ femb,
                            half_t* __restrict__ roleA)
{
    int n = BB * RR * 256;
    for (int i = blockIdx.x * blockDim.x + threadIdx.x; i < n; i += gridDim.x * blockDim.x) {
        int row = i / 256, c = i % 256;
        float v = (c < RDD) ? remb[(long long)voc[row] * RDD + c]
                            : femb[(long long)frames[row] * RDD + (c - RDD)];
        roleA[i] = (half_t)v;
    }
}

__global__ void k_repack_mapped(const half_t* __restrict__ mM, half_t* __restrict__ mh)
{
    long long n = (long long)BB * RPAD * DHS;
    for (long long i = (long long)blockIdx.x * blockDim.x + threadIdx.x; i < n;
         i += (long long)gridDim.x * blockDim.x) {
        int c  = (int)(i % DHS);
        int rr = (int)((i / DHS) % RPAD);
        int b  = (int)(i / ((long long)DHS * RPAD));
        mh[i] = (rr < RR) ? mM[((long long)b * RR + rr) * DHS + c] : (half_t)0.0f;
    }
}

__global__ void k_softmax_srl(const float* __restrict__ tag, const float* __restrict__ mask,
                              const int* __restrict__ targets, float* __restrict__ out,
                              float* __restrict__ acc)
{
    for (int row = blockIdx.x * blockDim.x + threadIdx.x; row < BL;
         row += gridDim.x * blockDim.x) {
        int b = row >> 7;
        float v[RR];
        float m = -1e30f;
#pragma unroll
        for (int r = 0; r < RR; ++r) {
            v[r] = tag[(long long)row * RPAD + r] + (mask[b * RR + r] - 1.0f) * 1.0e6f;
            m = fmaxf(m, v[r]);
        }
        float s = 0.0f;
#pragma unroll
        for (int r = 0; r < RR; ++r) s += expf(v[r] - m);
        float lse = m + logf(s);
#pragma unroll
        for (int r = 0; r < RR; ++r)
            out[3 + (long long)row * RR + r] = expf(v[r] - lse);
        int t = targets[row];
        if (t != 0) {
            atomicAdd(&acc[0], lse - v[t]);
            atomicAdd(&acc[1], 1.0f);
        }
    }
}

__global__ void k_dep_loss(const float* __restrict__ dep, const int* __restrict__ gold,
                           float* __restrict__ acc)
{
    for (int row = blockIdx.x * blockDim.x + threadIdx.x; row < BL;
         row += gridDim.x * blockDim.x) {
        float m = -1e30f; int am = 0;
        for (int c = 0; c < 50; ++c) {
            float x = dep[(long long)row * NDEP + c];
            if (x > m) { m = x; am = c; }
        }
        float s = 0.0f;
        for (int c = 0; c < 50; ++c) s += expf(dep[(long long)row * NDEP + c] - m);
        float lse = m + logf(s);
        int g = gold[row];
        if (g != 0) {
            atomicAdd(&acc[2], lse - dep[(long long)row * NDEP + g]);
            atomicAdd(&acc[3], 1.0f);
            if (am != g) atomicAdd(&acc[4], 1.0f);
        }
    }
}

__global__ void k_finalize(const float* __restrict__ acc, float* __restrict__ out)
{
    float srl = acc[0] / fmaxf(acc[1], 1.0f);
    float dep = acc[2] / fmaxf(acc[3], 1.0f);
    out[0] = srl; out[1] = dep; out[2] = srl + dep;
    out[3 + BL * RR]     = acc[4];
    out[3 + BL * RR + 1] = acc[3];
}

// ---------------------------------------------------------------------------
// Host driver
// ---------------------------------------------------------------------------
static inline void gemm(hipStream_t s,
                        const half_t* A, int lda, long long sA,
                        const half_t* W, int ldw, long long sW,
                        const float* bias,
                        const float* Cin, int ldci, long long sCi,
                        float* Co, int ldc, long long sC,
                        half_t* Ho, int ldh, long long sH,
                        int M, int N, int K, int act, int batch)
{
    const int tilesN = N / 16;
    const int nFull  = tilesN / 4;
    const int rem    = tilesN % 4;
    if (nFull) {
        int tiles = (M / 16) * nFull;
        dim3 g((tiles + 7) / 8, batch);
        hipLaunchKernelGGL(k_gemm<4>, g, dim3(256), 0, s,
                           A, lda, sA, W, ldw, sW, bias, Cin, ldci, sCi,
                           Co, ldc, sC, Ho, ldh, sH, M, K, act, 0, nFull);
    }
    if (rem) {
        int tiles = M / 16;
        dim3 g((tiles + 7) / 8, batch);
        int nbBase = nFull * 64;
        if (rem == 1)
            hipLaunchKernelGGL(k_gemm<1>, g, dim3(256), 0, s,
                               A, lda, sA, W, ldw, sW, bias, Cin, ldci, sCi,
                               Co, ldc, sC, Ho, ldh, sH, M, K, act, nbBase, 1);
        else if (rem == 2)
            hipLaunchKernelGGL(k_gemm<2>, g, dim3(256), 0, s,
                               A, lda, sA, W, ldw, sW, bias, Cin, ldci, sCi,
                               Co, ldc, sC, Ho, ldh, sH, M, K, act, nbBase, 1);
        else
            hipLaunchKernelGGL(k_gemm<3>, g, dim3(256), 0, s,
                               A, lda, sA, W, ldw, sW, bias, Cin, ldci, sCi,
                               Co, ldc, sC, Ho, ldh, sH, M, K, act, nbBase, 1);
    }
}

static inline dim3 gs(long long n) {
    long long b = (n + 255) / 256;
    if (b > 4096) b = 4096;
    return dim3((unsigned)b);
}

extern "C" void kernel_launch(void* const* d_in, const int* in_sizes, int n_in,
                              void* d_out, int out_size, void* d_ws, size_t ws_size,
                              hipStream_t stream)
{
    (void)in_sizes; (void)n_in; (void)out_size;
    // ---- inputs (setup_inputs order; params flattened in JAX sorted-key order)
    const int* sentence  = (const int*)d_in[0];
    const int* p_sent    = (const int*)d_in[1];
    const int* pos_tags  = (const int*)d_in[2];
    const int* tgt_idx   = (const int*)d_in[4];
    const float* region  = (const float*)d_in[5];
    const int* roles_voc = (const int*)d_in[6];
    const int* frames    = (const int*)d_in[7];
    const float* rmask   = (const float*)d_in[8];
    const int* lemma_idx = (const int*)d_in[9];
    const int* dep_tags  = (const int*)d_in[10];
    const int* dep_heads = (const int*)d_in[11];
    const int* targets   = (const int*)d_in[12];

    const float* frame_emb = (const float*)d_in[13];
    const float* h2tH_b = (const float*)d_in[14];
    const float* h2tH_w = (const float*)d_in[15];
    const float* h2tM_b = (const float*)d_in[16];
    const float* h2tM_w = (const float*)d_in[17];
    // lstm_share[l][d]: b,whh,wih at 18+(l*2+d)*3 ; lstm_srl at 30+
    const float* mlp_b  = (const float*)d_in[42];
    const float* mlp_w  = (const float*)d_in[43];
    const float* plemma = (const float*)d_in[44];
    const float* pos_e  = (const float*)d_in[45];
    const float* role_e = (const float*)d_in[46];
    const float* rmap_b = (const float*)d_in[47];
    const float* rmap_w = (const float*)d_in[48];
    const float* word_e = (const float*)d_in[49];
    const float* word_f = (const float*)d_in[50];

    float* out = (float*)d_out;

    // ---- workspace layout
    size_t off = 0;
    auto alloc = [&](size_t bytes) -> char* {
        char* p = (char*)d_ws + off;
        off += (bytes + 255) & ~(size_t)255;
        return p;
    };
    half_t* Xh    = (half_t*)alloc((size_t)BL * KX * 2);
    half_t* wih16[8];
    wih16[0] = (half_t*)alloc((size_t)H4 * KX * 2);      // share L0 d0
    wih16[1] = (half_t*)alloc((size_t)H4 * KX * 2);      // share L0 d1
    for (int i = 2; i < 8; ++i) wih16[i] = (half_t*)alloc((size_t)H4 * D2 * 2);
    half_t* whh16[8];
    for (int i = 0; i < 8; ++i) whh16[i] = (half_t*)alloc((size_t)H4 * KWH * 2);
    half_t* t0    = (half_t*)alloc((size_t)BL * D2 * 2);
    half_t* bfe   = (half_t*)alloc((size_t)BL * D2 * 2);
    half_t* h3h   = (half_t*)alloc((size_t)BL * D2 * 2);
    half_t* ce    = (half_t*)alloc((size_t)BL * D2 * 2);
    half_t* th16  = (half_t*)alloc((size_t)BL * D2 * 2);
    half_t* WM16  = (half_t*)alloc((size_t)D2 * D2 * 2);
    half_t* WH16  = (half_t*)alloc((size_t)D2 * D2 * 2);
    half_t* Wmlp  = (half_t*)alloc((size_t)NDEP * D2 * 2);
    half_t* Wrole = (half_t*)alloc((size_t)DHS * 256 * 2);
    half_t* hsh   = (half_t*)alloc((size_t)BL * DHS * 2);
    half_t* roleA = (half_t*)alloc((size_t)BB * RR * 256 * 2);
    half_t* mapM  = (half_t*)alloc((size_t)BB * RR * DHS * 2);
    half_t* maph  = (half_t*)alloc((size_t)BB * RPAD * DHS * 2);
    float*  xpF   = (float*)alloc((size_t)BL * H4 * 4);
    float*  xpB   = (float*)alloc((size_t)BL * H4 * 4);
    float*  tmf   = (float*)alloc((size_t)BL * D2 * 4);
    float*  depf  = (float*)alloc((size_t)BL * NDEP * 4);
    float*  tagf  = (float*)alloc((size_t)BB * LL * RPAD * 4);
    float*  bM    = (float*)alloc(D2 * 4);
    float*  bH    = (float*)alloc(D2 * 4);
    float*  bmlp  = (float*)alloc(NDEP * 4);
    float*  brole = (float*)alloc(DHS * 4);
    float*  acc   = (float*)alloc(8 * 4);
    (void)ws_size;

    hipMemsetAsync(d_ws, 0, off, stream);

    // ---- weight conversions (f32 -> f16, zero-padded)
    hipLaunchKernelGGL(k_cvt, gs((long long)H4 * KX), dim3(256), 0, stream,
                       (const float*)d_in[20], H4, 917, wih16[0], KX, H4);
    hipLaunchKernelGGL(k_cvt, gs((long long)H4 * KX), dim3(256), 0, stream,
                       (const float*)d_in[23], H4, 917, wih16[1], KX, H4);
    const int wih_src[6] = {26, 29, 32, 35, 38, 41};
    for (int i = 0; i < 6; ++i)
        hipLaunchKernelGGL(k_cvt, gs((long long)H4 * D2), dim3(256), 0, stream,
                           (const float*)d_in[wih_src[i]], H4, 600, wih16[2 + i], D2, H4);
    const int whh_src[8] = {19, 22, 25, 28, 31, 34, 37, 40};
    for (int i = 0; i < 8; ++i)
        hipLaunchKernelGGL(k_cvt, gs((long long)H4 * KWH), dim3(256), 0, stream,
                           (const float*)d_in[whh_src[i]], H4, 300, whh16[i], KWH, H4);
    hipLaunchKernelGGL(k_cvt, gs((long long)D2 * D2), dim3(256), 0, stream,
                       h2tM_w, 600, 600, WM16, D2, D2);
    hipLaunchKernelGGL(k_cvt, gs((long long)D2 * D2), dim3(256), 0, stream,
                       h2tH_w, 600, 600, WH16, D2, D2);
    hipLaunchKernelGGL(k_cvt, gs((long long)NDEP * D2), dim3(256), 0, stream,
                       mlp_w, 50, 600, Wmlp, D2, NDEP);
    hipLaunchKernelGGL(k_cvt, gs((long long)DHS * 256), dim3(256), 0, stream,
                       rmap_w, H4, 256, Wrole, 256, DHS);
    hipLaunchKernelGGL(k_padf32, dim3(4), dim3(256), 0, stream, h2tM_b, 600, bM, D2);
    hipLaunchKernelGGL(k_padf32, dim3(4), dim3(256), 0, stream, h2tH_b, 600, bH, D2);
    hipLaunchKernelGGL(k_padf32, dim3(1), dim3(256), 0, stream, mlp_b, 50, bmlp, NDEP);
    hipLaunchKernelGGL(k_padf32, dim3(8), dim3(256), 0, stream, rmap_b, H4, brole, DHS);

    // ---- embedding concat
    hipLaunchKernelGGL(k_embed, gs((long long)BL * KX), dim3(256), 0, stream,
                       sentence, p_sent, pos_tags, lemma_idx, region,
                       word_e, word_f, pos_e, plemma, Xh);

    // ---- BiLSTM stack: share L0/L1, srl L0/L1
    const half_t* layer_in[4] = {Xh, t0, bfe, t0};
    half_t*       layer_out[4] = {t0, bfe, t0, h3h};
    const int     layer_K[4] = {KX, D2, D2, D2};
    for (int ld = 0; ld < 4; ++ld) {
        int base = 18 + ld * 6;   // params: b,whh,wih per dir, share then srl
        const float* b0 = (const float*)d_in[base];
        const float* b1 = (const float*)d_in[base + 3];
        // input projections (batched over all 2048 tokens)
        gemm(stream, layer_in[ld], layer_K[ld], 0, wih16[ld*2+0], layer_K[ld], 0,
             b0, nullptr, 0, 0, xpF, H4, 0, nullptr, 0, 0, BL, H4, layer_K[ld], 0, 1);
        gemm(stream, layer_in[ld], layer_K[ld], 0, wih16[ld*2+1], layer_K[ld], 0,
             b1, nullptr, 0, 0, xpB, H4, 0, nullptr, 0, 0, BL, H4, layer_K[ld], 0, 1);
        // sequential scan, both directions in parallel (2 persistent blocks)
        hipLaunchKernelGGL(k_lstm_scan, dim3(2), dim3(1024), 0, stream,
                           xpF, xpB, whh16[ld*2+0], whh16[ld*2+1], layer_out[ld], D2);
        // dep branch hangs off bf_e (after layer 1); do it before t0 is reused
        if (ld == 1) {
            hipLaunchKernelGGL(k_gather_dep, gs((long long)BL * D2), dim3(256), 0, stream,
                               bfe, dep_heads, ce);
            gemm(stream, bfe, D2, 0, WM16, D2, 0, bM, nullptr, 0, 0,
                 tmf, D2, 0, nullptr, 0, 0, BL, D2, D2, 0, 1);
            gemm(stream, ce, D2, 0, WH16, D2, 0, bH, tmf, D2, 0,
                 nullptr, 0, 0, th16, D2, 0, BL, D2, D2, 2, 1);        // tanh
            gemm(stream, th16, D2, 0, Wmlp, D2, 0, bmlp, nullptr, 0, 0,
                 depf, NDEP, 0, nullptr, 0, 0, BL, NDEP, D2, 0, 1);
        }
    }

    // ---- SRL head
    hipLaunchKernelGGL(k_predcat, gs((long long)BL * DHS), dim3(256), 0, stream,
                       h3h, tgt_idx, hsh);
    hipLaunchKernelGGL(k_rolebuild, gs((long long)BB * RR * 256), dim3(256), 0, stream,
                       roles_voc, frames, role_e, frame_emb, roleA);
    gemm(stream, roleA, 256, 0, Wrole, 256, 0, brole, nullptr, 0, 0,
         nullptr, 0, 0, mapM, DHS, 0, BB * RR, DHS, 256, 1, 1);        // relu
    hipLaunchKernelGGL(k_repack_mapped, gs((long long)BB * RPAD * DHS), dim3(256), 0, stream,
                       mapM, maph);
    // tag_space: per-batch (128 x 1216) x (1216 x 32)
    gemm(stream, hsh, DHS, (long long)LL * DHS, maph, DHS, (long long)RPAD * DHS,
         nullptr, nullptr, 0, 0, tagf, RPAD, (long long)LL * RPAD,
         nullptr, 0, 0, LL, RPAD, DHS, 0, BB);

    // ---- losses / probs
    hipLaunchKernelGGL(k_softmax_srl, dim3(16), dim3(128), 0, stream,
                       tagf, rmask, targets, out, acc);
    hipLaunchKernelGGL(k_dep_loss, dim3(16), dim3(128), 0, stream,
                       depf, dep_tags, acc);
    hipLaunchKernelGGL(k_finalize, dim3(1), dim3(1), 0, stream, acc, out);
}
